// MultiHeadAttentionHierarchicalCls_65300682768913
// MI455X (gfx1250) — matrix operand order repaired and checked by reading, not compiled, not run on hardware
//
#include <hip/hip_runtime.h>
#include <hip/hip_bf16.h>
#include <math.h>

// ---------------------------------------------------------------------------
// MI455X (gfx1250) implementation of MultiHeadAttentionHierarchicalCls.
// All large GEMMs + attention run on v_wmma_f32_16x16x32_bf16 (wave32).
// f32 inputs are converted to bf16 (RNE) when staged into LDS fragments;
// accumulation, softmax, LayerNorm, pooling and routing stay in f32.
// ---------------------------------------------------------------------------

#define BATCH  256
#define NTOK   196
#define INDIM  1024
#define DMODEL 256
#define NHEAD  8
#define HDIM   32
#define NCLS   1000
#define SUB0   10
#define SUB1   20
#define NROWS  (BATCH * NTOK)   // 50176

typedef __attribute__((ext_vector_type(16))) __bf16        bf16x16;
typedef __attribute__((ext_vector_type(8)))  float         floatx8;
typedef __attribute__((ext_vector_type(8)))  unsigned short ushortx8;

union BFrag {
  bf16x16       v;
  ushortx8      h[2];
  unsigned short u[16];
};

__device__ __forceinline__ unsigned short f2bf(float x) {
  unsigned int u = __float_as_uint(x);
  u += 0x7FFFu + ((u >> 16) & 1u);          // round-to-nearest-even
  return (unsigned short)(u >> 16);
}

__device__ __forceinline__ float gelu_tanh(float x) {
  const float c = 0.7978845608028654f;      // sqrt(2/pi)
  float x3 = x * x * x;
  return 0.5f * x * (1.0f + tanhf(c * (x + 0.044715f * x3)));
}

// ---------------------------------------------------------------------------
// Generic bf16-WMMA GEMM: C[M,N] = act(A[M,K] * W[N,K]^T + bias[N])
// Tile: 128(M) x 64(N) per 256-thread block (8 waves). K-step = 32.
// Wave w computes rows [16w,16w+16) x all 64 cols -> 4 accumulators.
// If A2 != nullptr the A operand is the row-normalized concat
//   ((k < K1 ? A[m,k] : A2[m,k-K1]) - rmean[m]) * rrstd[m]
// (fuses the 1280-wide LayerNorm into the feature-selection GEMM).
// ---------------------------------------------------------------------------
__global__ __launch_bounds__(256) void gemm_bf16_wmma(
    const float* __restrict__ A,  const float* __restrict__ A2,
    const float* __restrict__ rmean, const float* __restrict__ rrstd,
    const float* __restrict__ W,  const float* __restrict__ bias,
    float* __restrict__ C, int M, int N, int K, int K1, int act)
{
  __shared__ __attribute__((aligned(16))) unsigned short As[128 * 32]; // 8 KB
  __shared__ __attribute__((aligned(16))) unsigned short Bs[64 * 32];  // 4 KB

  const int t    = threadIdx.x;
  const int wv   = t >> 5;
  const int lane = t & 31;
  const int r    = lane & 15;
  const int hl   = lane >> 4;
  const int m0   = blockIdx.x * 128;
  const int n0   = blockIdx.y * 64;

  floatx8 acc[4];
  #pragma unroll
  for (int j = 0; j < 4; ++j)
    #pragma unroll
    for (int e = 0; e < 8; ++e) acc[j][e] = 0.0f;

  const int  arow = t >> 1;            // 0..127  (A tile row)
  const int  ach  = (t & 1) << 4;      // 0 or 16 (K half)
  const int  gm   = m0 + arow;
  const bool has2 = (A2 != nullptr);
  const int  K2   = K - K1;
  float mv = 0.0f, rs = 1.0f;
  if (has2 && gm < M) { mv = rmean[gm]; rs = rrstd[gm]; }

  const int nkt = K >> 5;
  for (int kt = 0; kt < nkt; ++kt) {
    const int k0 = kt << 5;

    // ---- stage A tile (128 x 32) as bf16 ----
    {
      float tmp[16];
      if (gm < M) {
        const int kk = k0 + ach;
        if (!has2) {
          const float* p = A + (size_t)gm * K + kk;
          if (kt + 1 < nkt) __builtin_prefetch((const void*)(p + 32), 0, 1);
          #pragma unroll
          for (int i = 0; i < 16; ++i) tmp[i] = p[i];
        } else {
          const float* p = (kk < K1) ? (A  + (size_t)gm * K1 + kk)
                                     : (A2 + (size_t)gm * K2 + (kk - K1));
          #pragma unroll
          for (int i = 0; i < 16; ++i) tmp[i] = (p[i] - mv) * rs;
        }
      } else {
        #pragma unroll
        for (int i = 0; i < 16; ++i) tmp[i] = 0.0f;
      }
      #pragma unroll
      for (int i = 0; i < 16; ++i) As[arow * 32 + ach + i] = f2bf(tmp[i]);
    }

    // ---- stage W tile (64 x 32) as bf16 ----
    if (t < 128) {
      const int wrow = t >> 1;
      const int wch  = (t & 1) << 4;
      const int gn   = n0 + wrow;
      float tmp[16];
      if (gn < N) {
        const float* p = W + (size_t)gn * K + k0 + wch;
        if (kt + 1 < nkt) __builtin_prefetch((const void*)(p + 32), 0, 1);
        #pragma unroll
        for (int i = 0; i < 16; ++i) tmp[i] = p[i];
      } else {
        #pragma unroll
        for (int i = 0; i < 16; ++i) tmp[i] = 0.0f;
      }
      #pragma unroll
      for (int i = 0; i < 16; ++i) Bs[wrow * 32 + wch + i] = f2bf(tmp[i]);
    }
    __syncthreads();

    // ---- WMMA: A frag (rows 16*wv..+15), 4 B frags (cols 0..63) ----
    BFrag a;
    const int ar = (wv << 4) + r;
    a.h[0] = *(const ushortx8*)&As[ar * 32 + (hl << 3)];
    a.h[1] = *(const ushortx8*)&As[ar * 32 + 16 + (hl << 3)];
    #pragma unroll
    for (int j = 0; j < 4; ++j) {
      BFrag b;
      const int br = (j << 4) + r;
      b.h[0] = *(const ushortx8*)&Bs[br * 32 + (hl << 3)];
      b.h[1] = *(const ushortx8*)&Bs[br * 32 + 16 + (hl << 3)];
      acc[j] = __builtin_amdgcn_wmma_f32_16x16x32_bf16(
          false, a.v, false, b.v, (short)0, acc[j], false, false);
    }
    __syncthreads();
  }

  // ---- epilogue: bias + activation + store ----
  #pragma unroll
  for (int j = 0; j < 4; ++j) {
    const int col = n0 + (j << 4) + r;
    if (col >= N) continue;
    const float bvv = bias ? bias[col] : 0.0f;
    #pragma unroll
    for (int e = 0; e < 8; ++e) {
      const int grow = m0 + (wv << 4) + e + (hl << 3);
      if (grow < M) {
        float x = acc[j][e] + bvv;
        if (act == 1) x = gelu_tanh(x);
        C[(size_t)grow * N + col] = x;
      }
    }
  }
}

// ---------------------------------------------------------------------------
// Multi-head cross attention: one workgroup per (batch, head).
// K (208x32) and V^T (32x224) staged in LDS as bf16, zero padded.
// Per 16-query chunk: QK^T via WMMA (HD=32 == one K-step), f32 softmax,
// P converted to bf16, then P*V via WMMA.  q/k/v/ao layout: [B,N,H,HD].
// ---------------------------------------------------------------------------
__global__ __launch_bounds__(256) void attn_kernel(
    const float* __restrict__ qg, const float* __restrict__ kg,
    const float* __restrict__ vg, float* __restrict__ aog)
{
  __shared__ __attribute__((aligned(16))) unsigned short Ks[208 * 32];   // 13.0 KB
  __shared__ __attribute__((aligned(16))) unsigned short VsT[32 * 224];  // 14.0 KB
  __shared__ float Ssc[16 * 208];                                        // 13.0 KB
  __shared__ __attribute__((aligned(16))) unsigned short Pp[16 * 224];   //  7.0 KB

  const int t    = threadIdx.x;
  const int wv   = t >> 5;
  const int lane = t & 31;
  const int r    = lane & 15;
  const int hl   = lane >> 4;
  const int bh   = blockIdx.x;
  const int b    = bh >> 3;
  const int hh   = bh & 7;
  const float scale = 0.17677669529663687f;   // HD^-0.5

  // stage K (row-major keys x 32) and V^T (32 x keys), zero padded
  for (int e = t; e < 208 * 32; e += 256) {
    const int key = e >> 5, d = e & 31;
    float val = 0.0f;
    if (key < NTOK) val = kg[(((size_t)b * NTOK + key) * NHEAD + hh) * HDIM + d];
    Ks[e] = f2bf(val);
  }
  for (int e = t; e < 32 * 224; e += 256) {
    const int d = e / 224, key = e % 224;
    float val = 0.0f;
    if (key < NTOK) val = vg[(((size_t)b * NTOK + key) * NHEAD + hh) * HDIM + d];
    VsT[e] = f2bf(val);
  }
  __syncthreads();

  for (int q0 = 0; q0 < NTOK; q0 += 16) {
    // Q fragment directly from global (rows are contiguous 32 floats)
    BFrag aq;
    {
      const int qi = q0 + r;
      float tmp[16];
      if (qi < NTOK) {
        const float* p = qg + (((size_t)b * NTOK + qi) * NHEAD + hh) * HDIM;
        #pragma unroll
        for (int i = 0; i < 8; ++i) tmp[i]     = p[(hl << 3) + i];
        #pragma unroll
        for (int i = 0; i < 8; ++i) tmp[8 + i] = p[16 + (hl << 3) + i];
      } else {
        #pragma unroll
        for (int i = 0; i < 16; ++i) tmp[i] = 0.0f;
      }
      #pragma unroll
      for (int i = 0; i < 16; ++i) aq.u[i] = f2bf(tmp[i]);
    }

    // S = Q * K^T  (13 key tiles distributed over 8 waves)
    for (int j = wv; j < 13; j += 8) {
      BFrag bk;
      const int kr = (j << 4) + r;
      bk.h[0] = *(const ushortx8*)&Ks[kr * 32 + (hl << 3)];
      bk.h[1] = *(const ushortx8*)&Ks[kr * 32 + 16 + (hl << 3)];
      floatx8 sa;
      #pragma unroll
      for (int e = 0; e < 8; ++e) sa[e] = 0.0f;
      sa = __builtin_amdgcn_wmma_f32_16x16x32_bf16(
          false, aq.v, false, bk.v, (short)0, sa, false, false);
      #pragma unroll
      for (int e = 0; e < 8; ++e)
        Ssc[(e + (hl << 3)) * 208 + (j << 4) + r] = sa[e];
    }
    __syncthreads();

    // softmax over the 196 valid keys (one thread per query row)
    if (t < 16) {
      const float* srow = &Ssc[t * 208];
      float mx = -1e30f;
      for (int c = 0; c < NTOK; ++c) mx = fmaxf(mx, srow[c] * scale);
      float sum = 0.0f;
      for (int c = 0; c < NTOK; ++c) sum += __expf(srow[c] * scale - mx);
      const float inv = 1.0f / sum;
      unsigned short* prow = &Pp[t * 224];
      for (int c = 0; c < NTOK; ++c) prow[c] = f2bf(__expf(srow[c] * scale - mx) * inv);
      for (int c = NTOK; c < 224; ++c) prow[c] = 0;
    }
    __syncthreads();

    // O = P * V  (waves 0,1 each own 16 of the 32 head dims, 7 K-chunks)
    if (wv < 2) {
      floatx8 oa;
      #pragma unroll
      for (int e = 0; e < 8; ++e) oa[e] = 0.0f;
      #pragma unroll
      for (int kc = 0; kc < 7; ++kc) {
        BFrag ap, bv2;
        ap.h[0] = *(const ushortx8*)&Pp[r * 224 + (kc << 5) + (hl << 3)];
        ap.h[1] = *(const ushortx8*)&Pp[r * 224 + (kc << 5) + 16 + (hl << 3)];
        const int vr = (wv << 4) + r;
        bv2.h[0] = *(const ushortx8*)&VsT[vr * 224 + (kc << 5) + (hl << 3)];
        bv2.h[1] = *(const ushortx8*)&VsT[vr * 224 + (kc << 5) + 16 + (hl << 3)];
        oa = __builtin_amdgcn_wmma_f32_16x16x32_bf16(
            false, ap.v, false, bv2.v, (short)0, oa, false, false);
      }
      #pragma unroll
      for (int e = 0; e < 8; ++e) {
        const int qi = q0 + e + (hl << 3);
        if (qi < NTOK)
          aog[(((size_t)b * NTOK + qi) * NHEAD + hh) * HDIM + (wv << 4) + r] = oa[e];
      }
    }
    __syncthreads();
  }
}

// ---------------------------------------------------------------------------
// Per-row mean/rstd over concat(h[1024], integ[256]); one wave32 per row.
// ---------------------------------------------------------------------------
__global__ __launch_bounds__(256) void rowstats_kernel(
    const float* __restrict__ hin, const float* __restrict__ integ,
    float* __restrict__ rmean, float* __restrict__ rrstd)
{
  const int t = threadIdx.x;
  const int wv = t >> 5, lane = t & 31;
  const int row = blockIdx.x * 8 + wv;
  if (row >= NROWS) return;
  const float* ph = hin   + (size_t)row * INDIM;
  const float* pi = integ + (size_t)row * DMODEL;
  float s = 0.0f, s2 = 0.0f;
  for (int i = lane; i < INDIM; i += 32)  { float x = ph[i]; s += x; s2 += x * x; }
  for (int i = lane; i < DMODEL; i += 32) { float x = pi[i]; s += x; s2 += x * x; }
  #pragma unroll
  for (int off = 16; off > 0; off >>= 1) {
    s  += __shfl_xor(s,  off);
    s2 += __shfl_xor(s2, off);
  }
  if (lane == 0) {
    const float m = s * (1.0f / 1280.0f);
    const float var = s2 * (1.0f / 1280.0f) - m * m;
    rmean[row] = m;
    rrstd[row] = rsqrtf(var + 1e-5f);
  }
}

// ---------------------------------------------------------------------------
// Token pooling: pooled = mean_n fs, sp = mean_n se   (grid=B, block=256=D)
// ---------------------------------------------------------------------------
__global__ __launch_bounds__(256) void pool_kernel(
    const float* __restrict__ fs, const float* __restrict__ se,
    float* __restrict__ pooled, float* __restrict__ sp)
{
  const int b = blockIdx.x, d = threadIdx.x;
  const float* pf = fs + (size_t)b * NTOK * DMODEL + d;
  const float* ps = se + (size_t)b * NTOK * DMODEL + d;
  float s1 = 0.0f, s2 = 0.0f;
  for (int n = 0; n < NTOK; ++n) { s1 += pf[n * DMODEL]; s2 += ps[n * DMODEL]; }
  pooled[b * DMODEL + d] = s1 * (1.0f / NTOK);
  sp[b * DMODEL + d]     = s2 * (1.0f / NTOK);
}

// ---------------------------------------------------------------------------
// LayerNorm over 1000 parent logits + argmax routing (LN and softmax are
// monotonic per row, so argmax of pre-LN logits == reference routing).
// ---------------------------------------------------------------------------
__global__ __launch_bounds__(256) void ln_argmax_kernel(
    const float* __restrict__ logits, float* __restrict__ out, int* __restrict__ yhat)
{
  __shared__ float red[256], red2[256], rbv[256];
  __shared__ int   ridx[256];
  __shared__ float sm, sr;
  const int b = blockIdx.x, t = threadIdx.x;
  const float* row = logits + (size_t)b * NCLS;
  float s = 0.0f, s2 = 0.0f, best = -1e30f;
  int bi = 0;
  for (int c = t; c < NCLS; c += 256) {
    float x = row[c];
    s += x; s2 += x * x;
    if (x > best) { best = x; bi = c; }
  }
  red[t] = s; red2[t] = s2; rbv[t] = best; ridx[t] = bi;
  __syncthreads();
  for (int off = 128; off > 0; off >>= 1) {
    if (t < off) {
      red[t]  += red[t + off];
      red2[t] += red2[t + off];
      if (rbv[t + off] > rbv[t] ||
          (rbv[t + off] == rbv[t] && ridx[t + off] < ridx[t])) {
        rbv[t] = rbv[t + off]; ridx[t] = ridx[t + off];
      }
    }
    __syncthreads();
  }
  if (t == 0) {
    const float m = red[0] * (1.0f / NCLS);
    const float var = red2[0] * (1.0f / NCLS) - m * m;
    sm = m; sr = rsqrtf(var + 1e-5f);
    yhat[b] = ridx[0];
  }
  __syncthreads();
  for (int c = t; c < NCLS; c += 256)
    out[(size_t)b * NCLS + c] = (row[c] - sm) * sr;
}

// ---------------------------------------------------------------------------
// Routed per-sample expert matvecs + LayerNorm (one wave32 per sample).
// ---------------------------------------------------------------------------
__global__ __launch_bounds__(32) void child_kernel(
    const float* __restrict__ sp, const int* __restrict__ yhat,
    const float* __restrict__ Wc0, const float* __restrict__ bc0,
    const float* __restrict__ Wc1, const float* __restrict__ bc1,
    float* __restrict__ c0out, float* __restrict__ c1out)
{
  __shared__ float spv[DMODEL];
  __shared__ float vals[SUB1];
  __shared__ float mS, rS;
  const int b = blockIdx.x, t = threadIdx.x;
  const int e = yhat[b];
  for (int d = t; d < DMODEL; d += 32) spv[d] = sp[(size_t)b * DMODEL + d];
  __syncthreads();

  // child level 0 (10 outputs)
  if (t < SUB0) {
    float acc = bc0[(size_t)e * SUB0 + t];
    const float* w = Wc0 + (size_t)e * DMODEL * SUB0 + t;
    for (int d = 0; d < DMODEL; ++d) acc += spv[d] * w[d * SUB0];
    vals[t] = acc;
  }
  __syncthreads();
  if (t == 0) {
    float s = 0.0f;
    for (int i = 0; i < SUB0; ++i) s += vals[i];
    const float m = s / SUB0;
    float s2 = 0.0f;
    for (int i = 0; i < SUB0; ++i) { float d = vals[i] - m; s2 += d * d; }
    mS = m; rS = rsqrtf(s2 / SUB0 + 1e-5f);
  }
  __syncthreads();
  if (t < SUB0) c0out[(size_t)b * SUB0 + t] = (vals[t] - mS) * rS;
  __syncthreads();

  // child level 1 (20 outputs)
  if (t < SUB1) {
    float acc = bc1[(size_t)e * SUB1 + t];
    const float* w = Wc1 + (size_t)e * DMODEL * SUB1 + t;
    for (int d = 0; d < DMODEL; ++d) acc += spv[d] * w[d * SUB1];
    vals[t] = acc;
  }
  __syncthreads();
  if (t == 0) {
    float s = 0.0f;
    for (int i = 0; i < SUB1; ++i) s += vals[i];
    const float m = s / SUB1;
    float s2 = 0.0f;
    for (int i = 0; i < SUB1; ++i) { float d = vals[i] - m; s2 += d * d; }
    mS = m; rS = rsqrtf(s2 / SUB1 + 1e-5f);
  }
  __syncthreads();
  if (t < SUB1) c1out[(size_t)b * SUB1 + t] = (vals[t] - mS) * rS;
}

// ---------------------------------------------------------------------------
extern "C" void kernel_launch(void* const* d_in, const int* in_sizes, int n_in,
                              void* d_out, int out_size, void* d_ws, size_t ws_size,
                              hipStream_t stream) {
  (void)in_sizes; (void)n_in; (void)out_size; (void)ws_size;

  const float* h   = (const float*)d_in[0];
  const float* Wp  = (const float*)d_in[1];
  const float* bp  = (const float*)d_in[2];
  const float* Ws  = (const float*)d_in[3];
  const float* bs  = (const float*)d_in[4];
  const float* Wq  = (const float*)d_in[5];
  const float* bq  = (const float*)d_in[6];
  const float* Wk  = (const float*)d_in[7];
  const float* bk  = (const float*)d_in[8];
  const float* Wv  = (const float*)d_in[9];
  const float* bv  = (const float*)d_in[10];
  const float* Wo  = (const float*)d_in[11];
  const float* bo  = (const float*)d_in[12];
  const float* Wfs = (const float*)d_in[13];
  const float* bfs = (const float*)d_in[14];
  const float* Wpc = (const float*)d_in[15];
  const float* bpc = (const float*)d_in[16];
  const float* Wc0 = (const float*)d_in[17];
  const float* bc0 = (const float*)d_in[18];
  const float* Wc1 = (const float*)d_in[19];
  const float* bc1 = (const float*)d_in[20];

  // outputs (concat in return order): parent_logits, c0, c1, fs, se
  float* out        = (float*)d_out;
  float* parent_out = out;
  float* c0_out     = parent_out + (size_t)BATCH * NCLS;
  float* c1_out     = c0_out + (size_t)BATCH * SUB0;
  float* fs_out     = c1_out + (size_t)BATCH * SUB1;
  float* se_out     = fs_out + (size_t)NROWS * DMODEL;

  // workspace layout
  float* ws = (float*)d_ws;
  const size_t SZ = (size_t)NROWS * DMODEL;       // 12,845,056 floats
  float* pe     = ws;
  float* qbuf   = ws + 1 * SZ;
  float* kbuf   = ws + 2 * SZ;
  float* vbuf   = ws + 3 * SZ;
  float* aobuf  = ws + 4 * SZ;
  float* integ  = ws + 5 * SZ;
  float* rmean  = ws + 6 * SZ;
  float* rrstd  = rmean + NROWS;
  float* pooled = rrstd + NROWS;
  float* spool  = pooled + (size_t)BATCH * DMODEL;
  float* logits = spool + (size_t)BATCH * DMODEL;
  int*   yhat   = (int*)(logits + (size_t)BATCH * NCLS);

  const dim3 blk(256);
  const dim3 gBig(NROWS / 128, DMODEL / 64);      // 392 x 4

  // pe = gelu(h Wp^T + bp), se = gelu(h Ws^T + bs)
  gemm_bf16_wmma<<<gBig, blk, 0, stream>>>(h, nullptr, nullptr, nullptr,
      Wp, bp, pe, NROWS, DMODEL, INDIM, INDIM, 1);
  gemm_bf16_wmma<<<gBig, blk, 0, stream>>>(h, nullptr, nullptr, nullptr,
      Ws, bs, se_out, NROWS, DMODEL, INDIM, INDIM, 1);

  // q/k/v projections
  gemm_bf16_wmma<<<gBig, blk, 0, stream>>>(pe, nullptr, nullptr, nullptr,
      Wq, bq, qbuf, NROWS, DMODEL, DMODEL, DMODEL, 0);
  gemm_bf16_wmma<<<gBig, blk, 0, stream>>>(se_out, nullptr, nullptr, nullptr,
      Wk, bk, kbuf, NROWS, DMODEL, DMODEL, DMODEL, 0);
  gemm_bf16_wmma<<<gBig, blk, 0, stream>>>(se_out, nullptr, nullptr, nullptr,
      Wv, bv, vbuf, NROWS, DMODEL, DMODEL, DMODEL, 0);

  // attention -> ao
  attn_kernel<<<BATCH * NHEAD, blk, 0, stream>>>(qbuf, kbuf, vbuf, aobuf);

  // integ = ao Wo^T + bo
  gemm_bf16_wmma<<<gBig, blk, 0, stream>>>(aobuf, nullptr, nullptr, nullptr,
      Wo, bo, integ, NROWS, DMODEL, DMODEL, DMODEL, 0);

  // per-row LN stats over concat(h, integ)
  rowstats_kernel<<<NROWS / 8, blk, 0, stream>>>(h, integ, rmean, rrstd);

  // fs = LN(concat(h, integ)) Wfs^T + bfs   (LN fused into A operand)
  gemm_bf16_wmma<<<gBig, blk, 0, stream>>>(h, integ, rmean, rrstd,
      Wfs, bfs, fs_out, NROWS, DMODEL, INDIM + DMODEL, INDIM, 0);

  // token pooling
  pool_kernel<<<BATCH, blk, 0, stream>>>(fs_out, se_out, pooled, spool);

  // parent logits (M=256, N=1000 -> 16 N-tiles with bounds checks)
  gemm_bf16_wmma<<<dim3(BATCH / 128, (NCLS + 63) / 64), blk, 0, stream>>>(
      pooled, nullptr, nullptr, nullptr, Wpc, bpc, logits,
      BATCH, NCLS, DMODEL, DMODEL, 0);

  // LN + argmax routing
  ln_argmax_kernel<<<BATCH, blk, 0, stream>>>(logits, parent_out, yhat);

  // routed expert classifiers + LN
  child_kernel<<<BATCH, dim3(32), 0, stream>>>(spool, yhat,
      Wc0, bc0, Wc1, bc1, c0_out, c1_out);
}
